// TDNN_11793980195222
// MI455X (gfx1250) — matrix operand order repaired and checked by reading, not compiled, not run on hardware
//
#include <hip/hip_runtime.h>
#include <hip/hip_bf16.h>

// ---------------------------------------------------------------------------
// TDNN (B=32,T=1024,Din=512,Dout=512,KW=5) as an implicit-im2col f16 GEMM
// using v_wmma_f32_16x16x32_f16 on gfx1250 (wave32).
//
//   out[b,o,t] = ReLU( sum_k  Wh[o,k] * Xh[b, t*512 + k]  + bias[o] ),
//   k = w*512 + i, K = 2560, T_out = 1020.
//
// Because k orders (w outer, i inner), each im2col column (b,t) is a
// CONTIGUOUS 2560-halfword slice of Xh starting at row t. No gather needed.
//
// Workspace layout in d_ws (requires ~34.5 MiB):
//   [0)                Wh : f16, 512 x 2560, row-major (k = w*512+i)
//   [2,621,440 bytes)  Xh : f16, 32*1024*512 (same layout as x)
// ---------------------------------------------------------------------------

typedef __attribute__((ext_vector_type(16))) _Float16 v16h;
typedef __attribute__((ext_vector_type(8)))  _Float16 v8h;
typedef __attribute__((ext_vector_type(4)))  _Float16 v4h;
typedef __attribute__((ext_vector_type(8)))  float    v8f;
typedef __attribute__((ext_vector_type(4)))  float    v4f;

#define TDNN_B     32
#define TDNN_T     1024
#define TDNN_DIN   512
#define TDNN_DOUT  512
#define TDNN_KW    5
#define TDNN_TOUT  1020          // T - KW + 1
#define TDNN_K     2560          // DIN * KW

// ---------------- pre-pass 1: x fp32 -> f16 (bulk convert) ------------------
__global__ __launch_bounds__(256) void tdnn_cvt_x(const float* __restrict__ x,
                                                  _Float16* __restrict__ xh) {
  int i = blockIdx.x * 256 + threadIdx.x;          // i indexes float4 groups
  v4f f = ((const v4f* __restrict__)x)[i];
  ((v4h* __restrict__)xh)[i] = __builtin_convertvector(f, v4h);
}

// ------- pre-pass 2: W (o,i,w) fp32 -> Wh (o, w*512+i) f16 ------------------
__global__ __launch_bounds__(256) void tdnn_cvt_w(const float* __restrict__ w,
                                                  _Float16* __restrict__ wh) {
  int j = blockIdx.x * 256 + threadIdx.x;          // 512*2560 total, exact
  int o  = j / TDNN_K;
  int r  = j - o * TDNN_K;
  int kw = r >> 9;                                  // r / 512
  int ci = r & 511;                                 // r % 512
  wh[j] = (_Float16)w[(size_t)o * TDNN_K + ci * TDNN_KW + kw];
}

// ---------------- main kernel: implicit-im2col WMMA GEMM --------------------
// grid.x = B * 8  (8 N-tiles of 128 columns per batch; 8*128 >= 1020)
// grid.y = DOUT / 64 = 8 (M macro-tiles of 64)
// block  = 256 threads = 8 waves, arranged 2(M) x 4(N); wave tile = 32x32 C.
__global__ __launch_bounds__(256) void tdnn_wmma_gemm(
    const _Float16* __restrict__ wh,   // [512, 2560] row-major f16
    const _Float16* __restrict__ xh,   // [32*1024, 512] f16
    const float*    __restrict__ bias, // [512] f32
    float*          __restrict__ out)  // [32, 512, 1020] f32
{
  const int lane = threadIdx.x & 31;
  const int wave = threadIdx.x >> 5;        // 0..7
  const int wm   = wave & 1;                // 2 waves along M
  const int wn   = wave >> 1;               // 4 waves along N

  const int b   = blockIdx.x >> 3;          // batch
  const int nt  = blockIdx.x & 7;           // N macro-tile within batch
  const int t0  = nt * 128;

  const int mbase = blockIdx.y * 64 + wm * 32;   // 32 output rows for wave
  const int nbase = t0 + wn * 32;                // 32 output cols for wave

  const int lm = lane & 15;                 // row/col within 16
  const int hs = lane >> 4;                 // lane-half select

  // A fragments: lane<16 holds K 0..15 of row lm,
  // lane>=16 holds K 16..31 of row lm. Two 16-byte chunks per lane.
  const _Float16* arow0 = wh + (size_t)(mbase + lm) * TDNN_K + hs * 16;
  const _Float16* arow1 = arow0 + (size_t)16 * TDNN_K;

  // B fragments (contiguous k, same order as A): lanes 0-15 hold K 0..15 of col lm, lanes 16-31
  // hold K 16..31 of col lm -> one contiguous 32-byte chunk per lane, which is
  // exactly how the implicit-im2col column lies in Xh. Clamp tail columns.
  int tc0 = nbase + lm;        if (tc0 > TDNN_TOUT - 1) tc0 = TDNN_TOUT - 1;
  int tc1 = nbase + 16 + lm;   if (tc1 > TDNN_TOUT - 1) tc1 = TDNN_TOUT - 1;
  const _Float16* bcol0 = xh + ((size_t)(b * TDNN_T + tc0)) * TDNN_DIN + hs * 16;
  const _Float16* bcol1 = xh + ((size_t)(b * TDNN_T + tc1)) * TDNN_DIN + hs * 16;

  v8f acc[2][2] = {};

#pragma unroll 2
  for (int k0 = 0; k0 < TDNN_K; k0 += 32) {
    v8h a0lo = *(const v8h*)(arow0 + k0);
    v8h a0hi = *(const v8h*)(arow0 + k0 + 8);
    v8h a1lo = *(const v8h*)(arow1 + k0);
    v8h a1hi = *(const v8h*)(arow1 + k0 + 8);
    v16h a0 = __builtin_shufflevector(a0lo, a0hi,
                0,1,2,3,4,5,6,7,8,9,10,11,12,13,14,15);
    v16h a1 = __builtin_shufflevector(a1lo, a1hi,
                0,1,2,3,4,5,6,7,8,9,10,11,12,13,14,15);
    v16h b0 = *(const v16h*)(bcol0 + k0);
    v16h b1 = *(const v16h*)(bcol1 + k0);

    acc[0][0] = __builtin_amdgcn_wmma_f32_16x16x32_f16(
        false, a0, false, b0, (short)0, acc[0][0], false, false);
    acc[0][1] = __builtin_amdgcn_wmma_f32_16x16x32_f16(
        false, a0, false, b1, (short)0, acc[0][1], false, false);
    acc[1][0] = __builtin_amdgcn_wmma_f32_16x16x32_f16(
        false, a1, false, b0, (short)0, acc[1][0], false, false);
    acc[1][1] = __builtin_amdgcn_wmma_f32_16x16x32_f16(
        false, a1, false, b1, (short)0, acc[1][1], false, false);
  }

  // Epilogue: bias + ReLU + store. C/D layout: VGPR r of lane holds
  // M = (hs*8 + r), N = lm within the 16x16 subtile.
#pragma unroll
  for (int i = 0; i < 2; ++i) {
    const int row_base = mbase + i * 16 + hs * 8;
    float bv[8];
#pragma unroll
    for (int r = 0; r < 8; ++r) bv[r] = bias[row_base + r];
#pragma unroll
    for (int j = 0; j < 2; ++j) {
      const int t = nbase + j * 16 + lm;
      if (t < TDNN_TOUT) {
#pragma unroll
        for (int r = 0; r < 8; ++r) {
          float v = acc[i][j][r] + bv[r];
          out[((size_t)(b * TDNN_DOUT + row_base + r)) * TDNN_TOUT + t] =
              v > 0.f ? v : 0.f;
        }
      }
    }
  }
}

extern "C" void kernel_launch(void* const* d_in, const int* in_sizes, int n_in,
                              void* d_out, int out_size, void* d_ws, size_t ws_size,
                              hipStream_t stream) {
  (void)in_sizes; (void)n_in; (void)out_size; (void)ws_size;

  const float* x    = (const float*)d_in[0];   // [32,1024,512]
  const float* wker = (const float*)d_in[1];   // [512,512,5]
  const float* bias = (const float*)d_in[2];   // [512]
  float* out = (float*)d_out;                  // [32,512,1020]

  _Float16* wh = (_Float16*)d_ws;                                   // 2.5 MiB
  _Float16* xh = (_Float16*)((char*)d_ws + (size_t)TDNN_DOUT * TDNN_K * 2);

  // x: 32*1024*512 = 16,777,216 elems -> 4,194,304 float4 groups
  tdnn_cvt_x<<<4194304 / 256, 256, 0, stream>>>(x, xh);
  // W: 512*2560 = 1,310,720 elems
  tdnn_cvt_w<<<1310720 / 256, 256, 0, stream>>>(wker, wh);

  dim3 grid(TDNN_B * 8, TDNN_DOUT / 64);   // (256, 8)
  tdnn_wmma_gemm<<<grid, 256, 0, stream>>>(wh, xh, bias, out);
}
